// CrystalTransformer_28398323761889
// MI455X (gfx1250) — compile-verified, matrix operand order checked by reading, not run on hardware
//
#include <hip/hip_runtime.h>

#define N_NODES 20000
#define N_EDGES 320000
#define N_GRAPH 256
#define D_AT 92
#define D_AT_P 96    // padded to multiple of 32
#define D_EG 50
#define D_EG_P 64    // padded to multiple of 32
#define DIM 64
#define HEADS 4
#define HD 256       // HEADS * DIM
#define LAYERS 3

typedef __attribute__((ext_vector_type(8)))  __bf16 v8bf;
typedef __attribute__((ext_vector_type(16))) __bf16 v16bf;
typedef __attribute__((ext_vector_type(8)))  float  v8f;

// ---------------------------------------------------------------- utilities
__global__ void ct_fill_f32(float* __restrict__ p, float v, int n) {
  int i = blockIdx.x * blockDim.x + threadIdx.x;
  if (i < n) p[i] = v;
}

// A-side convert: out[m*Kp + k] = (k < K) ? (bf16)in[m*K + k] : 0
__global__ void ct_convert_pad_bf16(const float* __restrict__ in, __bf16* __restrict__ out,
                                    int M, int K, int Kp) {
  int i = blockIdx.x * blockDim.x + threadIdx.x;
  if (i >= M * Kp) return;
  int m = i / Kp, k = i - m * Kp;
  out[i] = (k < K) ? (__bf16)in[(size_t)m * K + k] : (__bf16)0.0f;
}

// B-side convert: W[K x Nc] -> Wt[Nc x Kp] (transposed + K padded)
__global__ void ct_transpose_w_bf16(const float* __restrict__ in, __bf16* __restrict__ out,
                                    int K, int Nc, int Kp) {
  int i = blockIdx.x * blockDim.x + threadIdx.x;
  if (i >= Nc * Kp) return;
  int n = i / Kp, k = i - n * Kp;
  out[i] = (k < K) ? (__bf16)in[(size_t)k * Nc + n] : (__bf16)0.0f;
}

__device__ inline void atomicMaxFloat(float* addr, float val) {
  if (val >= 0.0f)
    atomicMax((int*)addr, __float_as_int(val));
  else
    atomicMin((unsigned int*)addr, __float_as_uint(val));
}

// ---------------------------------------------------------------- WMMA GEMM
// C[M,Nc] = A[M,Kp](bf16, K padded) * Bt[Nc,Kp](bf16, transposed) + bias
// Block: 256 threads = 8 waves; tile 64x64; wave tile 16x32 (2 WMMA accs).
// Tiles staged in LDS via CDNA5 async-to-LDS DMA (ASYNCcnt).
#define TM 64
#define TN 64
#define TK 32
#define LDK (TK + 8)   // 80B rows: 16B aligned, bank-spread

__global__ __launch_bounds__(256)
void ct_gemm_bf16_wmma(const __bf16* __restrict__ A, const __bf16* __restrict__ Bt,
                       const float* __restrict__ bias, float* __restrict__ C,
                       int M, int Kp, int Nc)
{
  __shared__ __bf16 As[TM][LDK];   // [row][k]
  __shared__ __bf16 Bs[TN][LDK];   // [col][k]
  const int tid  = threadIdx.x;
  const int wave = tid >> 5;
  const int lane = tid & 31;
  const int bm = blockIdx.x * TM;
  const int bn = blockIdx.y * TN;
  const int wm = (wave & 3) * 16;
  const int wn = (wave >> 2) * 32;
  const int half = lane >> 4;   // 0: lanes 0-15, 1: lanes 16-31
  const int l16  = lane & 15;

  // per-thread 16B tile chunks (one async DMA each per K-step)
  const int ar = tid >> 2;            // A row in tile (0..63)
  const int ac = (tid & 3) * 8;       // A k-chunk start (0,8,16,24)
  int arow = bm + ar;
  if (arow >= M) arow = M - 1;        // clamp: OOB rows feed unstored C rows only
  const __bf16* a_gp = A + (size_t)arow * Kp + ac;
  const unsigned a_lds = (unsigned)(uintptr_t)&As[ar][ac];

  const int bc  = tid >> 2;           // B col in tile (0..63)
  const int bk8 = (tid & 3) * 8;      // k-chunk
  const __bf16* b_gp = Bt + (size_t)(bn + bc) * Kp + bk8;
  const unsigned b_lds = (unsigned)(uintptr_t)&Bs[bc][bk8];

  v8f acc0 = {}, acc1 = {};

  for (int k0 = 0; k0 < Kp; k0 += TK) {
    __syncthreads();  // LDS reuse fence (prev iteration consumers done)
    // CDNA5 async DMA: global -> LDS, 16B per lane, tracked by ASYNCcnt
    asm volatile("global_load_async_to_lds_b128 %0, %1, off"
                 :: "v"(a_lds), "v"((unsigned long long)(uintptr_t)(a_gp + k0))
                 : "memory");
    asm volatile("global_load_async_to_lds_b128 %0, %1, off"
                 :: "v"(b_lds), "v"((unsigned long long)(uintptr_t)(b_gp + k0))
                 : "memory");
    asm volatile("s_wait_asynccnt 0x0" ::: "memory");
    __syncthreads();

    // A fragment: lanes 0-15 row=wm+l16 K(0..7,16..23); lanes 16-31 K(8..15,24..31)
    v8bf alo = *(const v8bf*)&As[wm + l16][half * 8];
    v8bf ahi = *(const v8bf*)&As[wm + l16][16 + half * 8];
    v16bf afrag = __builtin_shufflevector(alo, ahi, 0,1,2,3,4,5,6,7,8,9,10,11,12,13,14,15);

    v8bf b0lo = *(const v8bf*)&Bs[wn + l16][half * 8];
    v8bf b0hi = *(const v8bf*)&Bs[wn + l16][16 + half * 8];
    v16bf bfrag0 = __builtin_shufflevector(b0lo, b0hi, 0,1,2,3,4,5,6,7,8,9,10,11,12,13,14,15);

    v8bf b1lo = *(const v8bf*)&Bs[wn + 16 + l16][half * 8];
    v8bf b1hi = *(const v8bf*)&Bs[wn + 16 + l16][16 + half * 8];
    v16bf bfrag1 = __builtin_shufflevector(b1lo, b1hi, 0,1,2,3,4,5,6,7,8,9,10,11,12,13,14,15);

    acc0 = __builtin_amdgcn_wmma_f32_16x16x32_bf16(false, afrag, false, bfrag0,
                                                   (short)0, acc0, false, false);
    acc1 = __builtin_amdgcn_wmma_f32_16x16x32_bf16(false, afrag, false, bfrag1,
                                                   (short)0, acc1, false, false);
  }

  // Epilogue. C layout: VGPR j -> M = j + 8*half, N = l16.
  // bias is always a valid pointer (zeros buffer when layer has no bias).
  const int col0 = bn + wn + l16;
  const int col1 = col0 + 16;
  const float b0 = bias[col0];
  const float b1 = bias[col1];
  const int row0 = bm + wm + half * 8;
  float* crow = C + (size_t)row0 * Nc;

  if (bm + TM <= M) {
    // fast path: whole 64-row tile in bounds (all but the last M-block)
#pragma unroll
    for (int j = 0; j < 8; ++j) {
      crow[(size_t)j * Nc + col0] = acc0[j] + b0;
      crow[(size_t)j * Nc + col1] = acc1[j] + b1;
    }
  } else {
#pragma unroll
    for (int j = 0; j < 8; ++j) {
      if (row0 + j < M) {
        crow[(size_t)j * Nc + col0] = acc0[j] + b0;
        crow[(size_t)j * Nc + col1] = acc1[j] + b1;
      }
    }
  }
}

// ---------------------------------------------------------------- edge phase
__global__ void ct_edge_alpha(const float* __restrict__ q, const float* __restrict__ k,
                              const float* __restrict__ ee, const int* __restrict__ src,
                              const int* __restrict__ dst, float* __restrict__ alpha,
                              float* __restrict__ maxv)
{
  int gid = blockIdx.x * blockDim.x + threadIdx.x;
  if (gid >= N_EDGES * HEADS) return;
  int e = gid >> 2, hh = gid & 3;
  int s = src[e], t = dst[e];
  const float4* qp = (const float4*)(q  + (size_t)t * HD + hh * DIM);
  const float4* kp = (const float4*)(k  + (size_t)s * HD + hh * DIM);
  const float4* ep = (const float4*)(ee + (size_t)e * HD + hh * DIM);
  float acc = 0.0f;
#pragma unroll 4
  for (int i = 0; i < DIM / 4; ++i) {
    float4 qv = qp[i], kv = kp[i], ev = ep[i];
    acc += qv.x * (kv.x + ev.x) + qv.y * (kv.y + ev.y)
         + qv.z * (kv.z + ev.z) + qv.w * (kv.w + ev.w);
  }
  acc *= 0.125f; // 1/sqrt(64)
  alpha[gid] = acc;
  atomicMaxFloat(&maxv[t * HEADS + hh], acc);
}

__global__ void ct_edge_exp(float* __restrict__ alpha, const float* __restrict__ maxv,
                            float* __restrict__ denom, const int* __restrict__ dst)
{
  int gid = blockIdx.x * blockDim.x + threadIdx.x;
  if (gid >= N_EDGES * HEADS) return;
  int e = gid >> 2, hh = gid & 3;
  int t = dst[e];
  float ex = __expf(alpha[gid] - maxv[t * HEADS + hh]);
  alpha[gid] = ex;
  atomicAdd(&denom[t * HEADS + hh], ex);
}

__global__ void ct_edge_scatter(const float* __restrict__ ex, const float* __restrict__ v,
                                const float* __restrict__ ee, const int* __restrict__ src,
                                const int* __restrict__ dst, float* __restrict__ agg)
{
  int gid = blockIdx.x * blockDim.x + threadIdx.x;
  if (gid >= N_EDGES * DIM) return;
  int e = gid >> 6, d = gid & 63;
  int s = src[e], t = dst[e];
  const float* vp = v  + (size_t)s * HD + d;
  const float* ep = ee + (size_t)e * HD + d;
  float*       ap = agg + (size_t)t * HD + d;
#pragma unroll
  for (int hh = 0; hh < HEADS; ++hh) {
    float w = ex[e * HEADS + hh];
    atomicAdd(ap + hh * DIM, w * (vp[hh * DIM] + ep[hh * DIM]));
  }
}

__global__ void ct_node_update(const float* __restrict__ agg, const float* __restrict__ denom,
                               const float* __restrict__ skip, float* __restrict__ hn)
{
  int gid = blockIdx.x * blockDim.x + threadIdx.x;
  if (gid >= N_NODES * DIM) return;
  int n = gid >> 6, d = gid & 63;
  float sum = 0.0f;
#pragma unroll
  for (int hh = 0; hh < HEADS; ++hh) {
    float dn = denom[n * HEADS + hh];
    float inv = dn > 0.0f ? 1.0f / dn : 0.0f;
    sum += agg[(size_t)n * HD + hh * DIM + d] * inv;
  }
  float o = sum * (1.0f / HEADS) + skip[gid];
  hn[gid] = o > 0.0f ? o : 0.0f;
}

// ---------------------------------------------------------------- pooling
__global__ void ct_pool_accum(const float* __restrict__ h, const int* __restrict__ batch,
                              float* __restrict__ pooled, float* __restrict__ counts)
{
  int gid = blockIdx.x * blockDim.x + threadIdx.x;
  if (gid >= N_NODES * DIM) return;
  int n = gid >> 6, d = gid & 63;
  int g = batch[n];
  atomicAdd(&pooled[g * DIM + d], h[gid]);
  if (d == 0) atomicAdd(&counts[g], 1.0f);
}

__global__ void ct_pool_final(const float* __restrict__ pooled, const float* __restrict__ counts,
                              const float* __restrict__ W_out, const float* __restrict__ b_out,
                              float* __restrict__ out)
{
  int g = blockIdx.x * blockDim.x + threadIdx.x;
  if (g >= N_GRAPH) return;
  float c = counts[g];
  c = c > 0.0f ? c : 1.0f;
  float acc = 0.0f;
  for (int d = 0; d < DIM; ++d) acc += pooled[g * DIM + d] * W_out[d];
  out[g] = acc / c + b_out[0];
}

// ---------------------------------------------------------------- launcher
extern "C" void kernel_launch(void* const* d_in, const int* in_sizes, int n_in,
                              void* d_out, int out_size, void* d_ws, size_t ws_size,
                              hipStream_t stream) {
  (void)in_sizes; (void)n_in; (void)out_size; (void)ws_size;
  const float* x      = (const float*)d_in[0];
  const int*   eidx   = (const int*)  d_in[1];
  const float* eattr  = (const float*)d_in[2];
  const int*   batch  = (const int*)  d_in[3];
  const float* W_atom = (const float*)d_in[4];
  const float* b_atom = (const float*)d_in[5];
  const float* W_edge = (const float*)d_in[6];
  const float* b_edge = (const float*)d_in[7];
  const float* Wq     = (const float*)d_in[8];
  const float* bq     = (const float*)d_in[9];
  const float* Wk     = (const float*)d_in[10];
  const float* bk     = (const float*)d_in[11];
  const float* Wv     = (const float*)d_in[12];
  const float* bv     = (const float*)d_in[13];
  const float* We     = (const float*)d_in[14];
  const float* Wskip  = (const float*)d_in[15];
  const float* bskip  = (const float*)d_in[16];
  const float* W_out  = (const float*)d_in[17];
  const float* b_out  = (const float*)d_in[18];
  float* out = (float*)d_out;

  const int* src = eidx;             // edge_index[0]
  const int* dst = eidx + N_EDGES;   // edge_index[1]

  // ---- bump allocator on d_ws
  char* ws = (char*)d_ws;
  size_t off = 0;
  auto alloc = [&](size_t bytes) -> void* {
    void* p = ws + off;
    off += (bytes + 255) & ~(size_t)255;
    return p;
  };

  float* hbuf   = (float*)alloc((size_t)N_NODES * DIM * 4);
  float* h2buf  = (float*)alloc((size_t)N_NODES * DIM * 4);
  float* ebuf   = (float*)alloc((size_t)N_EDGES * DIM * 4);
  float* qbuf   = (float*)alloc((size_t)N_NODES * HD * 4);
  float* kbuf   = (float*)alloc((size_t)N_NODES * HD * 4);
  float* vbuf   = (float*)alloc((size_t)N_NODES * HD * 4);
  float* skipb  = (float*)alloc((size_t)N_NODES * DIM * 4);
  float* eebuf  = (float*)alloc((size_t)N_EDGES * HD * 4);
  float* alphab = (float*)alloc((size_t)N_EDGES * HEADS * 4);
  float* maxv   = (float*)alloc((size_t)N_NODES * HEADS * 4);
  float* denom  = (float*)alloc((size_t)N_NODES * HEADS * 4);
  float* aggb   = (float*)alloc((size_t)N_NODES * HD * 4);
  float* pooled = (float*)alloc((size_t)N_GRAPH * DIM * 4);
  float* counts = (float*)alloc((size_t)N_GRAPH * 4);
  float* zerosb = (float*)alloc((size_t)HD * 4);       // zero "bias" for We GEMM

  __bf16* x_bf     = (__bf16*)alloc((size_t)N_NODES * D_AT_P * 2);
  __bf16* ea_bf    = (__bf16*)alloc((size_t)N_EDGES * D_EG_P * 2);
  __bf16* h_bf     = (__bf16*)alloc((size_t)N_NODES * DIM * 2);
  __bf16* e_bf     = (__bf16*)alloc((size_t)N_EDGES * DIM * 2);
  __bf16* Watom_t  = (__bf16*)alloc((size_t)DIM * D_AT_P * 2);          // [Nc=64][Kp=96]
  __bf16* Wedge_t  = (__bf16*)alloc((size_t)DIM * D_EG_P * 2);          // [64][64]
  __bf16* Wq_t     = (__bf16*)alloc((size_t)LAYERS * HD * DIM * 2);     // [256][64] per layer
  __bf16* Wk_t     = (__bf16*)alloc((size_t)LAYERS * HD * DIM * 2);
  __bf16* Wv_t     = (__bf16*)alloc((size_t)LAYERS * HD * DIM * 2);
  __bf16* We_t     = (__bf16*)alloc((size_t)LAYERS * HD * DIM * 2);
  __bf16* Wskip_t  = (__bf16*)alloc((size_t)LAYERS * DIM * DIM * 2);    // [64][64] per layer

  auto cdiv = [](int a, int b) { return (a + b - 1) / b; };
  const int T = 256;

  // ---- one-time conversions: A-side pad, B-side transpose(+pad)
  ct_fill_f32<<<1, T, 0, stream>>>(zerosb, 0.0f, HD);
  ct_convert_pad_bf16<<<cdiv(N_NODES * D_AT_P, T), T, 0, stream>>>(x, x_bf, N_NODES, D_AT, D_AT_P);
  ct_convert_pad_bf16<<<cdiv(N_EDGES * D_EG_P, T), T, 0, stream>>>(eattr, ea_bf, N_EDGES, D_EG, D_EG_P);
  ct_transpose_w_bf16<<<cdiv(DIM * D_AT_P, T), T, 0, stream>>>(W_atom, Watom_t, D_AT, DIM, D_AT_P);
  ct_transpose_w_bf16<<<cdiv(DIM * D_EG_P, T), T, 0, stream>>>(W_edge, Wedge_t, D_EG, DIM, D_EG_P);
  for (int l = 0; l < LAYERS; ++l) {
    ct_transpose_w_bf16<<<cdiv(HD * DIM, T), T, 0, stream>>>(
        Wq + (size_t)l * DIM * HD, Wq_t + (size_t)l * HD * DIM, DIM, HD, DIM);
    ct_transpose_w_bf16<<<cdiv(HD * DIM, T), T, 0, stream>>>(
        Wk + (size_t)l * DIM * HD, Wk_t + (size_t)l * HD * DIM, DIM, HD, DIM);
    ct_transpose_w_bf16<<<cdiv(HD * DIM, T), T, 0, stream>>>(
        Wv + (size_t)l * DIM * HD, Wv_t + (size_t)l * HD * DIM, DIM, HD, DIM);
    ct_transpose_w_bf16<<<cdiv(HD * DIM, T), T, 0, stream>>>(
        We + (size_t)l * DIM * HD, We_t + (size_t)l * HD * DIM, DIM, HD, DIM);
    ct_transpose_w_bf16<<<cdiv(DIM * DIM, T), T, 0, stream>>>(
        Wskip + (size_t)l * DIM * DIM, Wskip_t + (size_t)l * DIM * DIM, DIM, DIM, DIM);
  }

  // ---- embeddings (WMMA)
  ct_gemm_bf16_wmma<<<dim3(cdiv(N_NODES, TM), 1), T, 0, stream>>>(
      x_bf, Watom_t, b_atom, hbuf, N_NODES, D_AT_P, DIM);
  ct_gemm_bf16_wmma<<<dim3(cdiv(N_EDGES, TM), 1), T, 0, stream>>>(
      ea_bf, Wedge_t, b_edge, ebuf, N_EDGES, D_EG_P, DIM);
  ct_convert_pad_bf16<<<cdiv(N_EDGES * DIM, T), T, 0, stream>>>(ebuf, e_bf, N_EDGES, DIM, DIM);

  float* hcur = hbuf;
  float* hnext = h2buf;

  for (int l = 0; l < LAYERS; ++l) {
    ct_convert_pad_bf16<<<cdiv(N_NODES * DIM, T), T, 0, stream>>>(hcur, h_bf, N_NODES, DIM, DIM);

    // dense projections (WMMA bf16, K=Kp=64)
    ct_gemm_bf16_wmma<<<dim3(cdiv(N_NODES, TM), HD / TN), T, 0, stream>>>(
        h_bf, Wq_t + (size_t)l * HD * DIM, bq + (size_t)l * HD, qbuf, N_NODES, DIM, HD);
    ct_gemm_bf16_wmma<<<dim3(cdiv(N_NODES, TM), HD / TN), T, 0, stream>>>(
        h_bf, Wk_t + (size_t)l * HD * DIM, bk + (size_t)l * HD, kbuf, N_NODES, DIM, HD);
    ct_gemm_bf16_wmma<<<dim3(cdiv(N_NODES, TM), HD / TN), T, 0, stream>>>(
        h_bf, Wv_t + (size_t)l * HD * DIM, bv + (size_t)l * HD, vbuf, N_NODES, DIM, HD);
    ct_gemm_bf16_wmma<<<dim3(cdiv(N_NODES, TM), DIM / TN), T, 0, stream>>>(
        h_bf, Wskip_t + (size_t)l * DIM * DIM, bskip + (size_t)l * DIM, skipb, N_NODES, DIM, DIM);
    ct_gemm_bf16_wmma<<<dim3(cdiv(N_EDGES, TM), HD / TN), T, 0, stream>>>(
        e_bf, We_t + (size_t)l * HD * DIM, zerosb, eebuf, N_EDGES, DIM, HD);

    // segment softmax + scatter
    ct_fill_f32<<<cdiv(N_NODES * HEADS, T), T, 0, stream>>>(maxv, -INFINITY, N_NODES * HEADS);
    ct_fill_f32<<<cdiv(N_NODES * HEADS, T), T, 0, stream>>>(denom, 0.0f, N_NODES * HEADS);
    ct_fill_f32<<<cdiv(N_NODES * HD, T), T, 0, stream>>>(aggb, 0.0f, N_NODES * HD);

    ct_edge_alpha<<<cdiv(N_EDGES * HEADS, T), T, 0, stream>>>(
        qbuf, kbuf, eebuf, src, dst, alphab, maxv);
    ct_edge_exp<<<cdiv(N_EDGES * HEADS, T), T, 0, stream>>>(alphab, maxv, denom, dst);
    ct_edge_scatter<<<cdiv(N_EDGES * DIM, T), T, 0, stream>>>(
        alphab, vbuf, eebuf, src, dst, aggb);
    ct_node_update<<<cdiv(N_NODES * DIM, T), T, 0, stream>>>(aggb, denom, skipb, hnext);

    float* tmp = hcur; hcur = hnext; hnext = tmp;
  }

  // ---- graph mean pool + output head
  ct_fill_f32<<<cdiv(N_GRAPH * DIM, T), T, 0, stream>>>(pooled, 0.0f, N_GRAPH * DIM);
  ct_fill_f32<<<1, T, 0, stream>>>(counts, 0.0f, N_GRAPH);
  ct_pool_accum<<<cdiv(N_NODES * DIM, T), T, 0, stream>>>(hcur, batch, pooled, counts);
  ct_pool_final<<<1, T, 0, stream>>>(pooled, counts, W_out, b_out, out);
}
